// QuestionGenerator_18262200943095
// MI455X (gfx1250) — compile-verified, matrix operand order checked by reading, not run on hardware
//
#include <hip/hip_runtime.h>
#include <cstdint>
#include <cstddef>

// ---------------------------------------------------------------------------
// QuestionGenerator on gfx1250 (MI455X, wave32, WMMA bf16 -> f32 accum)
//
// B=32, S=256, T=32 (31 decode steps), V=32000, E=128, H=256, HD=512
// All matmuls register-block 4 N-tiles per wave to reuse the A fragment.
// ---------------------------------------------------------------------------

typedef __attribute__((ext_vector_type(16))) __bf16 v16bf;
typedef __attribute__((ext_vector_type(8)))  float  v8f;

union bfu { unsigned short u; __bf16 b; };

__device__ __forceinline__ unsigned short f32_to_bf16(float f) {
  union { float f; unsigned u; } v; v.f = f;
  unsigned r = v.u + 0x7fffu + ((v.u >> 16) & 1u);   // round-to-nearest-even
  return (unsigned short)(r >> 16);
}
__device__ __forceinline__ float bf16_to_f32(unsigned short h) {
  union { unsigned u; float f; } v; v.u = ((unsigned)h) << 16;
  return v.f;
}
__device__ __forceinline__ __bf16 us_to_bf(unsigned short x) { bfu t; t.u = x; return t.b; }
__device__ __forceinline__ float sigmoidf_(float x) { return 1.f / (1.f + __expf(-x)); }

// ---- WMMA fragment helpers (wave32, ISA 7.12.2 layouts) --------------------
// A fragment: 16x32 bf16 tile from row-major src (row stride = ld elements).
// lane = g*16 + m ; VGPR j holds K = (j/4)*16 + g*8 + (j%4)*2 , +1
__device__ __forceinline__ v16bf load_frag16x32(const unsigned short* __restrict__ src, int ld) {
  int lane = threadIdx.x & 31;
  int m = lane & 15;
  int g = lane >> 4;
  const unsigned short* row = src + (size_t)m * ld;
  v16bf f;
#pragma unroll
  for (int j = 0; j < 8; ++j) {
    int k0 = ((j >> 2) << 4) + (g << 3) + ((j & 3) << 1);
    f[2 * j]     = us_to_bf(row[k0]);
    f[2 * j + 1] = us_to_bf(row[k0 + 1]);
  }
  return f;
}
// C/D fragment: 16x16 f32. lane = g*16+n ; VGPR r holds C[g*8+r][n]
__device__ __forceinline__ v8f load_cfrag(const float* __restrict__ src, int ld) {
  int lane = threadIdx.x & 31;
  int n = lane & 15, g = lane >> 4;
  v8f c;
#pragma unroll
  for (int r = 0; r < 8; ++r) c[r] = src[(size_t)(g * 8 + r) * ld + n];
  return c;
}
__device__ __forceinline__ void store_cfrag(float* __restrict__ dst, int ld, v8f c) {
  int lane = threadIdx.x & 31;
  int n = lane & 15, g = lane >> 4;
#pragma unroll
  for (int r = 0; r < 8; ++r) dst[(size_t)(g * 8 + r) * ld + n] = c[r];
}

// ---------------------------------------------------------------------------
// Generic GEMM: C[M,N] = A[M,K] @ W[N,K]^T + bias, bf16 in / f32 out.
// One wave computes a 16x64 output strip (1 M-tile x 4 N-tiles) so the A
// fragment is loaded once per K-step and fed to 4 WMMAs. Requires N % 64 == 0,
// M % 16 == 0, K % 32 == 0 (true for every GEMM in this model).
// Optional output row remap (t,b)->(b,t) for the final logits GEMM.
// ---------------------------------------------------------------------------
__global__ void gemm_bf16(const unsigned short* __restrict__ A,
                          const unsigned short* __restrict__ W,
                          const float* __restrict__ bias,
                          float* __restrict__ C,
                          int M, int N, int K, int remapT, int remapB) {
  int wave = (int)(threadIdx.x >> 5);
  int ngrp = N >> 6;                                    // groups of 4 N-tiles
  int grp = (int)blockIdx.x * (int)(blockDim.x >> 5) + wave;
  int total = (M >> 4) * ngrp;
  if (grp >= total) return;                             // wave-uniform exit
  int tm = grp / ngrp, gn = grp - tm * ngrp;
  int tn0 = gn << 2;
  const unsigned short* Ar = A + (size_t)tm * 16 * K;
  const unsigned short* W0 = W + (size_t)(tn0 + 0) * 16 * K;
  const unsigned short* W1 = W + (size_t)(tn0 + 1) * 16 * K;
  const unsigned short* W2 = W + (size_t)(tn0 + 2) * 16 * K;
  const unsigned short* W3 = W + (size_t)(tn0 + 3) * 16 * K;
  v8f acc0 = {}, acc1 = {}, acc2 = {}, acc3 = {};
  for (int k = 0; k < K; k += 32) {
    if (k + 32 < K) __builtin_prefetch(Ar + k + 32, 0, 1);   // global_prefetch_b8
    v16bf a  = load_frag16x32(Ar + k, K);
    v16bf b0 = load_frag16x32(W0 + k, K);
    v16bf b1 = load_frag16x32(W1 + k, K);
    v16bf b2 = load_frag16x32(W2 + k, K);
    v16bf b3 = load_frag16x32(W3 + k, K);
    acc0 = __builtin_amdgcn_wmma_f32_16x16x32_bf16(false, a, false, b0, (short)0, acc0, false, false);
    acc1 = __builtin_amdgcn_wmma_f32_16x16x32_bf16(false, a, false, b1, (short)0, acc1, false, false);
    acc2 = __builtin_amdgcn_wmma_f32_16x16x32_bf16(false, a, false, b2, (short)0, acc2, false, false);
    acc3 = __builtin_amdgcn_wmma_f32_16x16x32_bf16(false, a, false, b3, (short)0, acc3, false, false);
  }
  int lane = threadIdx.x & 31;
  int n = lane & 15, g = lane >> 4;
  v8f accs[4] = {acc0, acc1, acc2, acc3};
#pragma unroll
  for (int j = 0; j < 4; ++j) {
    int col = (tn0 + j) * 16 + n;
    float bv = bias ? bias[col] : 0.f;
#pragma unroll
    for (int r = 0; r < 8; ++r) {
      int m = tm * 16 + g * 8 + r;
      int row = m;
      if (remapT > 0) row = (m % remapB) * remapT + (m / remapB);   // [t*B+b] -> [b*T+t]
      C[(size_t)row * N + col] = accs[j][r] + bv;
    }
  }
}

// ---------------------------------------------------------------------------
// Weight / activation prep kernels
// ---------------------------------------------------------------------------
__global__ void conv_slice_bf16(const float* __restrict__ src, unsigned short* __restrict__ dst,
                                int rows, int src_ld, int col0, int cols) {
  int idx = blockIdx.x * blockDim.x + threadIdx.x;
  if (idx >= rows * cols) return;
  int r = idx / cols, c = idx - r * cols;
  dst[idx] = f32_to_bf16(src[(size_t)r * src_ld + col0 + c]);
}

// Wc = Wd_ih[:,128:640] + Wd_hh  ->  bf16 [2048,512]
__global__ void build_wc(const float* __restrict__ Wdih, const float* __restrict__ Wdhh,
                         unsigned short* __restrict__ dst) {
  int idx = blockIdx.x * blockDim.x + threadIdx.x;
  if (idx >= 2048 * 512) return;
  int r = idx >> 9, c = idx & 511;
  dst[idx] = f32_to_bf16(Wdih[(size_t)r * 640 + 128 + c] + Wdhh[(size_t)r * 512 + c]);
}

// xcat bf16 [s][b][256] = concat(embed[ctx], marker_embed[mark])
__global__ void build_xcat(const int* __restrict__ cids, const int* __restrict__ mids,
                           const float* __restrict__ embed, const float* __restrict__ membed,
                           unsigned short* __restrict__ xcat) {
  int idx = blockIdx.x * blockDim.x + threadIdx.x;
  if (idx >= 8192 * 256) return;
  int e = idx & 255, sb = idx >> 8;
  int b = sb & 31, s = sb >> 5;
  float v = (e < 128) ? embed[(size_t)cids[b * 256 + s] * 128 + e]
                      : membed[(size_t)mids[b * 256 + s] * 128 + (e - 128)];
  xcat[idx] = f32_to_bf16(v);
}

// decoder teacher-forced inputs bf16 [t][b][128], t = 0..30
__global__ void build_decx(const int* __restrict__ qids, const float* __restrict__ embed,
                           unsigned short* __restrict__ dst) {
  int idx = blockIdx.x * blockDim.x + threadIdx.x;
  if (idx >= 992 * 128) return;
  int e = idx & 127, tb = idx >> 7;
  int b = tb & 31, t = tb >> 5;
  dst[idx] = f32_to_bf16(embed[(size_t)qids[b * 32 + t] * 128 + e]);
}

// enc_out bf16 [b][s][512] = concat(Hf[s][b], Hb[s][b])   (Hb stored un-reversed)
__global__ void build_enc_out(const float* __restrict__ Hf, const float* __restrict__ Hb,
                              unsigned short* __restrict__ eo) {
  int idx = blockIdx.x * blockDim.x + threadIdx.x;
  if (idx >= 32 * 256 * 512) return;
  int d = idx & 511, sb = idx >> 9;
  int s = sb & 255, b = sb >> 8;
  float v = (d < 256) ? Hf[((size_t)s * 32 + b) * 256 + d]
                      : Hb[((size_t)s * 32 + b) * 256 + (d - 256)];
  eo[idx] = f32_to_bf16(v);
}

__global__ void init_dec_state(const float* __restrict__ Hf, const float* __restrict__ Hb,
                               const float* __restrict__ cfin,
                               unsigned short* __restrict__ h_bf, float* __restrict__ c_st) {
  int idx = blockIdx.x * blockDim.x + threadIdx.x;
  if (idx >= 32 * 512) return;
  int b = idx >> 9, j = idx & 511;
  float h, c;
  if (j < 256) { h = Hf[((size_t)255 * 32 + b) * 256 + j]; c = cfin[b * 256 + j]; }
  else         { h = Hb[(size_t)b * 256 + (j - 256)];      c = cfin[32 * 256 + b * 256 + (j - 256)]; }
  h_bf[idx] = f32_to_bf16(h);
  c_st[idx] = c;
}

// ---------------------------------------------------------------------------
// Persistent encoder scan: 1 block/direction, 32 waves. Per step:
// gates[32,1024] = pre[s] + h @ W_hh^T (WMMA, h in LDS as bf16) -> LSTM cell.
// Each wave owns a 16x64 strip (fixed tm, 4 consecutive tn) -> A frag loaded
// once per K-step from LDS and reused by 4 WMMAs.
// ---------------------------------------------------------------------------
__global__ void __launch_bounds__(1024)
encoder_scan(const float* __restrict__ preF, const float* __restrict__ preB,
             const unsigned short* __restrict__ WF, const unsigned short* __restrict__ WB,
             float* __restrict__ HF, float* __restrict__ HB,
             float* __restrict__ gbufF, float* __restrict__ gbufB,
             float* __restrict__ cfin) {
  const int dir = blockIdx.x;
  const float* pre = dir ? preB : preF;
  const unsigned short* W = dir ? WB : WF;
  float* Hout = dir ? HB : HF;
  float* gbuf = dir ? gbufB : gbufF;

  __shared__ unsigned short h_bf[32 * 256];   // 16 KB (WMMA A operand)
  __shared__ float          c_s[32 * 256];    // 32 KB
  int tid = threadIdx.x;
  for (int i = tid; i < 32 * 256; i += 1024) { h_bf[i] = 0; c_s[i] = 0.f; }
  __syncthreads();

  int wave = tid >> 5;
  // 2 x 64 = 128 tiles; wave -> tiles [wave*4, wave*4+3]: same tm, 4 consecutive tn
  int tile0 = wave * 4;
  int tm = tile0 >> 6, tn0 = tile0 & 63;
  const unsigned short* Ah = h_bf + (size_t)tm * 16 * 256;
  const unsigned short* Wr0 = W + (size_t)(tn0 + 0) * 16 * 256;
  const unsigned short* Wr1 = W + (size_t)(tn0 + 1) * 16 * 256;
  const unsigned short* Wr2 = W + (size_t)(tn0 + 2) * 16 * 256;
  const unsigned short* Wr3 = W + (size_t)(tn0 + 3) * 16 * 256;

  for (int t = 0; t < 256; ++t) {
    int s = dir ? (255 - t) : t;                      // original time index
    const float* pre_s = pre + (size_t)s * 32 * 1024;
    const float* cb = pre_s + (size_t)tm * 16 * 1024 + tn0 * 16;
    v8f acc0 = load_cfrag(cb, 1024);
    v8f acc1 = load_cfrag(cb + 16, 1024);
    v8f acc2 = load_cfrag(cb + 32, 1024);
    v8f acc3 = load_cfrag(cb + 48, 1024);
#pragma unroll
    for (int k = 0; k < 256; k += 32) {
      v16bf a  = load_frag16x32(Ah + k, 256);
      v16bf b0 = load_frag16x32(Wr0 + k, 256);
      v16bf b1 = load_frag16x32(Wr1 + k, 256);
      v16bf b2 = load_frag16x32(Wr2 + k, 256);
      v16bf b3 = load_frag16x32(Wr3 + k, 256);
      acc0 = __builtin_amdgcn_wmma_f32_16x16x32_bf16(false, a, false, b0, (short)0, acc0, false, false);
      acc1 = __builtin_amdgcn_wmma_f32_16x16x32_bf16(false, a, false, b1, (short)0, acc1, false, false);
      acc2 = __builtin_amdgcn_wmma_f32_16x16x32_bf16(false, a, false, b2, (short)0, acc2, false, false);
      acc3 = __builtin_amdgcn_wmma_f32_16x16x32_bf16(false, a, false, b3, (short)0, acc3, false, false);
    }
    float* gb0 = gbuf + (size_t)tm * 16 * 1024 + tn0 * 16;
    store_cfrag(gb0, 1024, acc0);
    store_cfrag(gb0 + 16, 1024, acc1);
    store_cfrag(gb0 + 32, 1024, acc2);
    store_cfrag(gb0 + 48, 1024, acc3);
    __threadfence_block();
    __syncthreads();
    // LSTM cell (gate order i,f,g,o)
    for (int e = tid; e < 32 * 256; e += 1024) {
      int b = e >> 8, j = e & 255;
      const float* gb = gbuf + (size_t)b * 1024;
      float ig = sigmoidf_(gb[j]);
      float fg = sigmoidf_(gb[256 + j]);
      float gg = tanhf(gb[512 + j]);
      float og = sigmoidf_(gb[768 + j]);
      float cn = fg * c_s[e] + ig * gg;
      float hn = og * tanhf(cn);
      c_s[e] = cn;
      h_bf[e] = f32_to_bf16(hn);
      Hout[(size_t)s * 32 * 256 + e] = hn;
    }
    __threadfence_block();
    __syncthreads();
  }
  for (int e = tid; e < 32 * 256; e += 1024) cfin[(size_t)dir * 32 * 256 + e] = c_s[e];
}

// ---------------------------------------------------------------------------
// Decoder step: gates[32,2048] = dec_pre[t] + h @ (Wd_ih[:,E:]+Wd_hh)^T -> cell.
// 2 x 128 = 256 tiles; wave -> 8 tiles (fixed tm, 8 consecutive tn) processed
// as two 4-tile groups with A-fragment reuse.
// ---------------------------------------------------------------------------
__global__ void __launch_bounds__(1024)
decoder_step_cell(const float* __restrict__ dec_pre,          // [31][32][2048]
                  const unsigned short* __restrict__ Wc,      // [2048][512] bf16
                  unsigned short* __restrict__ h_bf,          // [32][512] bf16
                  float* __restrict__ c_st,                   // [32][512]
                  float* __restrict__ gbuf,                   // [32][2048]
                  unsigned short* __restrict__ hc_bf,         // [992][1024] bf16
                  int t) {
  int tid = threadIdx.x, wave = tid >> 5;
  const float* pre_s = dec_pre + (size_t)t * 32 * 2048;
  int tile0 = wave * 8;
  int tm = tile0 >> 7;
  const unsigned short* Ah = h_bf + (size_t)tm * 16 * 512;
#pragma unroll
  for (int grp = 0; grp < 2; ++grp) {
    int tn0 = (tile0 & 127) + grp * 4;
    const unsigned short* Wr0 = Wc + (size_t)(tn0 + 0) * 16 * 512;
    const unsigned short* Wr1 = Wc + (size_t)(tn0 + 1) * 16 * 512;
    const unsigned short* Wr2 = Wc + (size_t)(tn0 + 2) * 16 * 512;
    const unsigned short* Wr3 = Wc + (size_t)(tn0 + 3) * 16 * 512;
    const float* cb = pre_s + (size_t)tm * 16 * 2048 + tn0 * 16;
    v8f acc0 = load_cfrag(cb, 2048);
    v8f acc1 = load_cfrag(cb + 16, 2048);
    v8f acc2 = load_cfrag(cb + 32, 2048);
    v8f acc3 = load_cfrag(cb + 48, 2048);
    for (int k = 0; k < 512; k += 32) {
      v16bf a  = load_frag16x32(Ah + k, 512);
      v16bf b0 = load_frag16x32(Wr0 + k, 512);
      v16bf b1 = load_frag16x32(Wr1 + k, 512);
      v16bf b2 = load_frag16x32(Wr2 + k, 512);
      v16bf b3 = load_frag16x32(Wr3 + k, 512);
      acc0 = __builtin_amdgcn_wmma_f32_16x16x32_bf16(false, a, false, b0, (short)0, acc0, false, false);
      acc1 = __builtin_amdgcn_wmma_f32_16x16x32_bf16(false, a, false, b1, (short)0, acc1, false, false);
      acc2 = __builtin_amdgcn_wmma_f32_16x16x32_bf16(false, a, false, b2, (short)0, acc2, false, false);
      acc3 = __builtin_amdgcn_wmma_f32_16x16x32_bf16(false, a, false, b3, (short)0, acc3, false, false);
    }
    float* gb0 = gbuf + (size_t)tm * 16 * 2048 + tn0 * 16;
    store_cfrag(gb0, 2048, acc0);
    store_cfrag(gb0 + 16, 2048, acc1);
    store_cfrag(gb0 + 32, 2048, acc2);
    store_cfrag(gb0 + 48, 2048, acc3);
  }
  __threadfence_block();
  __syncthreads();
  for (int e = tid; e < 32 * 512; e += 1024) {
    int b = e >> 9, j = e & 511;
    const float* gb = gbuf + (size_t)b * 2048;
    float ig = sigmoidf_(gb[j]);
    float fg = sigmoidf_(gb[512 + j]);
    float gg = tanhf(gb[1024 + j]);
    float og = sigmoidf_(gb[1536 + j]);
    float cn = fg * c_st[e] + ig * gg;
    float hn = og * tanhf(cn);
    c_st[e] = cn;
    unsigned short hb = f32_to_bf16(hn);
    h_bf[e] = hb;
    hc_bf[((size_t)t * 32 + b) * 1024 + j] = hb;
  }
}

// ---------------------------------------------------------------------------
// Additive attention + context: one block per batch row.
// scores[s] = tanh(h_part + enc_proj[b,s]) . a2 + a2b  (+mask bias); softmax;
// ctx = sum_s attn[s] * enc_out[b,s,:] -> hc row [512:1024] (bf16)
// ---------------------------------------------------------------------------
__global__ void attention_ctx(const float* __restrict__ h_part,       // [32][256]
                              const float* __restrict__ enc_proj,     // [32][256][256]
                              const unsigned short* __restrict__ enc_out, // [32][256][512]
                              const int* __restrict__ mask,           // [32][256]
                              const float* __restrict__ a2,           // [256]
                              const float* __restrict__ a2b,          // [1]
                              unsigned short* __restrict__ hc_bf,     // [992][1024]
                              int t) {
  int b = blockIdx.x;
  int tid = threadIdx.x;      // 256 threads; one score each
  __shared__ float sc[256];
  __shared__ float red[256];

  {
    const float* ep = enc_proj + ((size_t)b * 256 + tid) * 256;
    const float* hp = h_part + (size_t)b * 256;
    float acc = 0.f;
    for (int d = 0; d < 256; ++d) acc += tanhf(hp[d] + ep[d]) * a2[d];
    acc += a2b[0];
    if (mask[b * 256 + tid] == 0) acc += -1e30f;
    sc[tid] = acc;
  }
  __syncthreads();
  red[tid] = sc[tid];
  __syncthreads();
  for (int o = 128; o > 0; o >>= 1) { if (tid < o) red[tid] = fmaxf(red[tid], red[tid + o]); __syncthreads(); }
  float mx = red[0];
  __syncthreads();
  float ex = __expf(sc[tid] - mx);
  sc[tid] = ex;
  red[tid] = ex;
  __syncthreads();
  for (int o = 128; o > 0; o >>= 1) { if (tid < o) red[tid] += red[tid + o]; __syncthreads(); }
  float inv = 1.f / red[0];
  __syncthreads();

  float c0 = 0.f, c1 = 0.f;
  const unsigned short* eb = enc_out + (size_t)b * 256 * 512;
  for (int s = 0; s < 256; ++s) {
    float w = sc[s];
    c0 += w * bf16_to_f32(eb[(size_t)s * 512 + tid * 2]);
    c1 += w * bf16_to_f32(eb[(size_t)s * 512 + tid * 2 + 1]);
  }
  size_t row = ((size_t)t * 32 + b) * 1024;
  hc_bf[row + 512 + tid * 2]     = f32_to_bf16(c0 * inv);
  hc_bf[row + 512 + tid * 2 + 1] = f32_to_bf16(c1 * inv);
}

// ---------------------------------------------------------------------------
extern "C" void kernel_launch(void* const* d_in, const int* in_sizes, int n_in,
                              void* d_out, int out_size, void* d_ws, size_t ws_size,
                              hipStream_t stream) {
  (void)in_sizes; (void)n_in; (void)out_size; (void)ws_size;
  const int*   cids   = (const int*)d_in[0];
  const int*   mids   = (const int*)d_in[1];
  const int*   qids   = (const int*)d_in[2];
  const int*   mask   = (const int*)d_in[3];
  const float* embed  = (const float*)d_in[4];
  const float* membed = (const float*)d_in[5];
  const float* Wf_ih  = (const float*)d_in[6];
  const float* Wf_hh  = (const float*)d_in[7];
  const float* bf_    = (const float*)d_in[8];
  const float* Wb_ih  = (const float*)d_in[9];
  const float* Wb_hh  = (const float*)d_in[10];
  const float* bb_    = (const float*)d_in[11];
  const float* Wd_ih  = (const float*)d_in[12];
  const float* Wd_hh  = (const float*)d_in[13];
  const float* bd_    = (const float*)d_in[14];
  const float* A1_w   = (const float*)d_in[15];
  const float* A1_b   = (const float*)d_in[16];
  const float* A2_w   = (const float*)d_in[17];
  const float* A2_b   = (const float*)d_in[18];
  const float* O_w    = (const float*)d_in[19];
  const float* O_b    = (const float*)d_in[20];
  float* out = (float*)d_out;

  char* ws = (char*)d_ws;
  size_t off = 0;
  auto take = [&](size_t bytes) -> char* {
    char* p = ws + off;
    off += (bytes + 255) & ~(size_t)255;
    return p;
  };

  // Region A: encoder pre-activations (phase 1) aliased with O_w bf16 (phase 2).
  // 2 * 32 MiB >= 62.5 MiB needed for O_w bf16.
  char* regionA = take(2ULL * 8192 * 1024 * 4);
  float* preF = (float*)regionA;
  float* preB = (float*)(regionA + (size_t)8192 * 1024 * 4);
  unsigned short* OwB = (unsigned short*)regionA;

  unsigned short* xcat = (unsigned short*)take((size_t)8192 * 256 * 2);
  unsigned short* decx = (unsigned short*)take((size_t)992 * 128 * 2);
  float* Hf            = (float*)take((size_t)8192 * 256 * 4);
  float* Hb            = (float*)take((size_t)8192 * 256 * 4);
  unsigned short* encO = (unsigned short*)take((size_t)8192 * 512 * 2);
  float* encP          = (float*)take((size_t)8192 * 256 * 4);
  float* decPre        = (float*)take((size_t)992 * 2048 * 4);
  unsigned short* WfihB = (unsigned short*)take((size_t)1024 * 256 * 2);
  unsigned short* WfhhB = (unsigned short*)take((size_t)1024 * 256 * 2);
  unsigned short* WbihB = (unsigned short*)take((size_t)1024 * 256 * 2);
  unsigned short* WbhhB = (unsigned short*)take((size_t)1024 * 256 * 2);
  unsigned short* WdxB  = (unsigned short*)take((size_t)2048 * 128 * 2);
  unsigned short* WcB   = (unsigned short*)take((size_t)2048 * 512 * 2);
  unsigned short* A1eB  = (unsigned short*)take((size_t)256 * 512 * 2);
  unsigned short* A1dB  = (unsigned short*)take((size_t)256 * 512 * 2);
  unsigned short* hcB   = (unsigned short*)take((size_t)992 * 1024 * 2);
  float* cfin           = (float*)take((size_t)2 * 32 * 256 * 4);
  unsigned short* hSt   = (unsigned short*)take((size_t)32 * 512 * 2);
  float* cSt            = (float*)take((size_t)32 * 512 * 4);
  float* gbF            = (float*)take((size_t)32 * 1024 * 4);
  float* gbB            = (float*)take((size_t)32 * 1024 * 4);
  float* gbD            = (float*)take((size_t)32 * 2048 * 4);
  float* hPart          = (float*)take((size_t)32 * 256 * 4);

  auto nb = [](long total, int tpb) { return (int)((total + tpb - 1) / tpb); };

  // ---- weight conversions (disjoint from region A) ----
  conv_slice_bf16<<<nb(1024L * 256, 256), 256, 0, stream>>>(Wf_ih, WfihB, 1024, 256, 0, 256);
  conv_slice_bf16<<<nb(1024L * 256, 256), 256, 0, stream>>>(Wf_hh, WfhhB, 1024, 256, 0, 256);
  conv_slice_bf16<<<nb(1024L * 256, 256), 256, 0, stream>>>(Wb_ih, WbihB, 1024, 256, 0, 256);
  conv_slice_bf16<<<nb(1024L * 256, 256), 256, 0, stream>>>(Wb_hh, WbhhB, 1024, 256, 0, 256);
  conv_slice_bf16<<<nb(2048L * 128, 256), 256, 0, stream>>>(Wd_ih, WdxB, 2048, 640, 0, 128);
  build_wc<<<nb(2048L * 512, 256), 256, 0, stream>>>(Wd_ih, Wd_hh, WcB);
  conv_slice_bf16<<<nb(256L * 512, 256), 256, 0, stream>>>(A1_w, A1eB, 256, 1024, 512, 512);
  conv_slice_bf16<<<nb(256L * 512, 256), 256, 0, stream>>>(A1_w, A1dB, 256, 1024, 0, 512);

  // ---- activations ----
  build_xcat<<<nb(8192L * 256, 256), 256, 0, stream>>>(cids, mids, embed, membed, xcat);
  build_decx<<<nb(992L * 128, 256), 256, 0, stream>>>(qids, embed, decx);

  // ---- hoisted input projections: pre = xcat @ W_ih^T + b ----
  // work-units = (M/16)*(N/64)
  gemm_bf16<<<nb(512L * 16, 8), 256, 0, stream>>>(xcat, WfihB, bf_, preF, 8192, 1024, 256, 0, 0);
  gemm_bf16<<<nb(512L * 16, 8), 256, 0, stream>>>(xcat, WbihB, bb_, preB, 8192, 1024, 256, 0, 0);

  // ---- bidirectional encoder scan (persistent, 1 WG per direction) ----
  encoder_scan<<<2, 1024, 0, stream>>>(preF, preB, WfhhB, WbhhB, Hf, Hb, gbF, gbB, cfin);

  // region A is free now -> convert O_w to bf16 (62.5 MiB; L2-resident for final GEMM)
  build_enc_out<<<nb(32L * 256 * 512, 256), 256, 0, stream>>>(Hf, Hb, encO);
  conv_slice_bf16<<<nb(32000L * 1024, 256), 256, 0, stream>>>(O_w, OwB, 32000, 1024, 0, 1024);

  // enc-side attention projection (no bias) + decoder input projection (+bd)
  gemm_bf16<<<nb(512L * 4, 8), 256, 0, stream>>>(encO, A1eB, nullptr, encP, 8192, 256, 512, 0, 0);
  gemm_bf16<<<nb(62L * 32, 8), 256, 0, stream>>>(decx, WdxB, bd_, decPre, 992, 2048, 128, 0, 0);
  init_dec_state<<<nb(32L * 512, 256), 256, 0, stream>>>(Hf, Hb, cfin, hSt, cSt);

  // ---- decoder scan: 31 sequential steps (logits deferred to one big GEMM) ----
  for (int t = 0; t < 31; ++t) {
    decoder_step_cell<<<1, 1024, 0, stream>>>(decPre, WcB, hSt, cSt, gbD, hcB, t);
    gemm_bf16<<<1, 256, 0, stream>>>(hSt, A1dB, A1_b, hPart, 32, 256, 512, 0, 0);
    attention_ctx<<<32, 256, 0, stream>>>(hPart, encP, encO, mask, A2_w, A2_b, hcB, t);
  }

  // ---- dominant GEMM: logits[992,32000] = hc @ O_w^T + O_b, remapped to [B,T-1,V] ----
  gemm_bf16<<<nb(62L * 500, 8), 256, 0, stream>>>(hcB, OwB, O_b, out, 992, 32000, 1024, 31, 32);
}
